// DGCF_64287070486722
// MI455X (gfx1250) — matrix operand order repaired and checked by reading, not compiled, not run on hardware
//
#include <hip/hip_runtime.h>
#include <hip/hip_bf16.h>
#include <cstdint>

// ---------------- CDNA5 types ----------------
typedef __attribute__((ext_vector_type(16))) _Float16 v16h;
typedef __attribute__((ext_vector_type(8)))  _Float16 v8h;
typedef __attribute__((ext_vector_type(8)))  float    v8f;
typedef __attribute__((ext_vector_type(4)))  unsigned u32x4;
typedef __attribute__((ext_vector_type(8)))  int      i32x8;
typedef __attribute__((ext_vector_type(4)))  int      i32x4;

#define EMBED  64
#define FACTOR 4
#define CHUNK  16
#define DT     64   // edges per scatter tile (TDM staged)

__device__ __forceinline__ const float* ego_row(const float* ue, const float* ie,
                                                int n_user, int node) {
  return (node < n_user) ? (ue + (size_t)node * EMBED)
                         : (ie + (size_t)(node - n_user) * EMBED);
}

// ---------------- TDM: 1-D dword tile -> LDS (gfx1250 Tensor Data Mover) ----
// lds_byte_off is relative to the workgroup LDS base (shared arrays in these
// kernels are the sole LDS allocation, so they start at offset 0).
// This toolchain exposes the 6-arg builtin variant.
__device__ __forceinline__ void tdm_load_dwords(unsigned lds_byte_off,
                                                const void* gsrc,
                                                int tile_n, int tensor_n) {
  unsigned long long ga = (unsigned long long)(uintptr_t)gsrc;
  // D# group0: count=1 | lds_addr | global_addr[56:0] | type=2 ("image")
  u32x4 g0 = { 1u,
               lds_byte_off,
               (unsigned)(ga & 0xFFFFFFFFull),
               (unsigned)(((ga >> 32) & 0x1FFFFFFull) | (2u << 30)) };
  // D# group1: data_size=4B; tensor_dim0=tensor_n; tensor_dim1=1;
  //            tile_dim0=tile_n; tile_dim1=1; dim0_stride=tensor_n
  i32x8 g1 = { (int)(2u << 16),
               (int)((tensor_n & 0xFFFF) << 16),
               (int)(((unsigned)tensor_n >> 16) | (1u << 16)),
               (int)((tile_n & 0xFFFF) << 16),
               1,
               tensor_n,
               0, 0 };
  i32x4 z4 = {0, 0, 0, 0};
  i32x8 z8 = {0, 0, 0, 0, 0, 0, 0, 0};
  __builtin_amdgcn_tensor_load_to_lds(g0, g1, z4, z4, z8, 0);
}

// ---------------- utility fill ----------------
__global__ void k_fill(float* p, float v, long long n) {
  long long i = (long long)blockIdx.x * blockDim.x + threadIdx.x;
  if (i < n) p[i] = v;
}

// ---------------- softmax over FACTOR per edge (fv,sc layout: [K][E]) ------
__global__ void k_softmax(const float* __restrict__ fv, float* __restrict__ sc, int E) {
  int e = blockIdx.x * blockDim.x + threadIdx.x;
  if (e >= E) return;
  size_t Es = (size_t)E;
  float a0 = fv[e], a1 = fv[Es + e], a2 = fv[2 * Es + e], a3 = fv[3 * Es + e];
  float mx = fmaxf(fmaxf(a0, a1), fmaxf(a2, a3));
  float p0 = expf(a0 - mx), p1 = expf(a1 - mx), p2 = expf(a2 - mx), p3 = expf(a3 - mx);
  float inv = 1.0f / (p0 + p1 + p2 + p3);
  sc[e] = p0 * inv; sc[Es + e] = p1 * inv; sc[2 * Es + e] = p2 * inv; sc[3 * Es + e] = p3 * inv;
}

// ---------------- colsum[k][head[e]] += scores[k][e] ------------------------
__global__ void k_colsum(const int* __restrict__ head, const float* __restrict__ sc,
                         float* __restrict__ colsum, int E, int N) {
  int e = blockIdx.x * blockDim.x + threadIdx.x;
  int k = blockIdx.y;
  if (e < E) atomicAdd(&colsum[(size_t)k * N + head[e]], sc[(size_t)k * E + e]);
}

// ---------------- d = rsqrt(colsum) in place --------------------------------
__global__ void k_invsqrt(float* d, long long n) {
  long long i = (long long)blockIdx.x * blockDim.x + threadIdx.x;
  if (i < n) d[i] = rsqrtf(fmaxf(d[i], 1e-24f));
}

// ---------------- msg scatter: femb[head] += w * ego[tail] ------------------
// 256 threads = 64 edges x 4 factors; edge indices staged via TDM into LDS.
__global__ void __launch_bounds__(256)
k_scatter_msg(const float* __restrict__ ue, const float* __restrict__ ie, int n_user,
              const int* __restrict__ head, const int* __restrict__ tail, int E, int N,
              const float* __restrict__ sc, const float* __restrict__ dcol,
              float* __restrict__ femb) {
  __shared__ int tile[2 * DT];  // sole LDS alloc: head @ byte 0, tail @ byte 256
  const int e0 = blockIdx.x * DT;
  const int rem = (E - e0 < DT) ? (E - e0) : DT;
  if (threadIdx.x < 32) {  // wave 0 drives the Tensor Data Mover
    tdm_load_dwords(0u, head + e0, DT, rem);
    tdm_load_dwords((unsigned)(DT * sizeof(int)), tail + e0, DT, rem);
    __builtin_amdgcn_s_wait_tensorcnt(0);
  }
  __syncthreads();

  const int el = threadIdx.x >> 2;   // 0..63
  const int k  = threadIdx.x & 3;    // factor
  const int e  = e0 + el;
  if (e >= E) return;
  const int h = tile[el];
  const int t = tile[DT + el];
  const float w = sc[(size_t)k * E + e] * dcol[(size_t)k * N + t];
  const float4* p4 = (const float4*)(ego_row(ue, ie, n_user, t) + k * CHUNK);
  float* dst = femb + (size_t)h * EMBED + k * CHUNK;
#pragma unroll
  for (int q = 0; q < 4; ++q) {
    float4 v = p4[q];
    atomicAdd(dst + 4 * q + 0, w * v.x);
    atomicAdd(dst + 4 * q + 1, w * v.y);
    atomicAdd(dst + 4 * q + 2, w * v.z);
    atomicAdd(dst + 4 * q + 3, w * v.w);
  }
}

// ---------------- femb[n, 16k+c] *= dcol[k][n] ------------------------------
__global__ void k_scale(float* __restrict__ femb, const float* __restrict__ dcol, int N) {
  long long i = (long long)blockIdx.x * blockDim.x + threadIdx.x;
  if (i >= (long long)N * EMBED) return;
  int n = (int)(i >> 6);
  int k = (int)((i & 63) >> 4);
  femb[i] *= dcol[(size_t)k * N + n];
}

// ---------------- fv[k][e] += dot(norm(femb[head]), tanh(norm(ego[tail]))) --
// One wave per 16-edge tile; per factor, one v_wmma_f32_16x16x32_f16 computes
// the 16x16 all-pairs dot matrix; we keep the diagonal. K 16..31 zero-padded.
// Fragment loads are contiguous vector LDS loads (ds_load_b128), no divergent
// per-element loads.
__global__ void __launch_bounds__(32)
k_update_fv(const float* __restrict__ ue, const float* __restrict__ ie, int n_user,
            const int* __restrict__ head, const int* __restrict__ tail, int E, int N,
            const float* __restrict__ femb, float* __restrict__ fv) {
  __shared__ __align__(32) _Float16 Hs[16][16];  // normalized head femb rows
  __shared__ __align__(32) _Float16 Gs[16][16];  // tanh(normalized tail ego)
  const int lane = threadIdx.x;
  const int m    = lane & 15;
  const int hi   = lane >> 4;       // 0: H producer, 1: G producer
  const int e0   = blockIdx.x * 16;
  int e = e0 + m; if (e >= E) e = E - 1;   // clamp (E%16==0 in practice)

  for (int k = 0; k < FACTOR; ++k) {
    const float* src = (hi == 0)
        ? (femb + (size_t)head[e] * EMBED + k * CHUNK)
        : (ego_row(ue, ie, n_user, tail[e]) + k * CHUNK);
    float x[16];
    float ss = 0.f;
    const float4* p4 = (const float4*)src;
#pragma unroll
    for (int q = 0; q < 4; ++q) {
      float4 v = p4[q];
      x[4 * q + 0] = v.x; x[4 * q + 1] = v.y; x[4 * q + 2] = v.z; x[4 * q + 3] = v.w;
      ss += v.x * v.x + v.y * v.y + v.z * v.z + v.w * v.w;
    }
    float inv = 1.0f / fmaxf(sqrtf(ss), 1e-12f);

    // pack 16 halves in registers, then two 16B vector stores to LDS
    v8h r0, r1;
    if (hi == 0) {
#pragma unroll
      for (int cc = 0; cc < 8; ++cc) {
        r0[cc] = (_Float16)(x[cc] * inv);
        r1[cc] = (_Float16)(x[cc + 8] * inv);
      }
      *(v8h*)(&Hs[m][0]) = r0;
      *(v8h*)(&Hs[m][8]) = r1;
    } else {
#pragma unroll
      for (int cc = 0; cc < 8; ++cc) {
        r0[cc] = (_Float16)tanhf(x[cc] * inv);
        r1[cc] = (_Float16)tanhf(x[cc + 8] * inv);
      }
      *(v8h*)(&Gs[m][0]) = r0;
      *(v8h*)(&Gs[m][8]) = r1;
    }
    __syncthreads();

    // A (16x32 f16): lane half 'hi' holds K = hi*8 .. hi*8+7 in elems 0..7;
    //                elems 8..15 correspond to K>=16 -> zero padding.
    v8h alo = *(const v8h*)(&Hs[m][hi * 8]);   // single ds_load_b128
    v16h a;
#pragma unroll
    for (int j = 0; j < 8; ++j) { a[j] = alo[j]; a[j + 8] = (_Float16)0.0f; }

    // B (32x16 f16): lane half 'hi' holds K = hi*8 .. hi*8+7 in elems 0..7;
    //                elems 8..15 correspond to K>=16 -> zero padding.
    v8h g = *(const v8h*)(&Gs[m][hi * 8]);     // single ds_load_b128
    v16h b;
#pragma unroll
    for (int j = 0; j < 8; ++j) { b[j] = g[j]; b[j + 8] = (_Float16)0.0f; }

    v8f acc = {};
    acc = __builtin_amdgcn_wmma_f32_16x16x32_f16(false, a, false, b,
                                                 (short)0, acc, false, false);

    // Diagonal: lanes 0-7 hold (M=r=lane,N=lane); lanes 24-31 hold (M=8+r=lane-16)
    float dval = 0.f;
    bool  has  = (lane < 8) || (lane >= 24);
#pragma unroll
    for (int r = 0; r < 8; ++r) {
      if (lane < 8)        { if (r == lane)      dval = acc[r]; }
      else if (lane >= 24) { if (r == lane - 24) dval = acc[r]; }
    }
    if (has) {
      int me = (lane < 8) ? lane : (lane - 16);
      if (e0 + me < E) fv[(size_t)k * E + e0 + me] += dval;
    }
    __syncthreads();  // protect Hs/Gs before next factor
  }
}

// ---------------- out = 0.5*(ego + femb) ------------------------------------
__global__ void k_out(const float* __restrict__ ue, const float* __restrict__ ie, int n_user,
                      const float* __restrict__ femb, float* __restrict__ out, int N) {
  long long i = (long long)blockIdx.x * blockDim.x + threadIdx.x;
  if (i >= (long long)N * EMBED) return;
  int n = (int)(i >> 6);
  int cc = (int)(i & 63);
  float egov = (n < n_user) ? ue[(size_t)n * EMBED + cc]
                            : ie[(size_t)(n - n_user) * EMBED + cc];
  out[i] = 0.5f * (egov + femb[i]);
}

// ---------------- host orchestration ----------------------------------------
extern "C" void kernel_launch(void* const* d_in, const int* in_sizes, int n_in,
                              void* d_out, int out_size, void* d_ws, size_t ws_size,
                              hipStream_t stream) {
  const float* ue  = (const float*)d_in[0];
  const float* ie  = (const float*)d_in[1];
  const int* head  = (const int*)d_in[2];
  const int* tail  = (const int*)d_in[3];
  const int n_user = in_sizes[0] / EMBED;
  const int n_item = in_sizes[1] / EMBED;
  const int N      = n_user + n_item;
  const int E      = in_sizes[2];

  char* ws = (char*)d_ws;
  auto al = [](size_t x) { return (x + 255) & ~(size_t)255; };
  size_t off = 0;
  float* fv   = (float*)(ws + off); off = al(off + (size_t)FACTOR * E * sizeof(float));
  float* sc   = (float*)(ws + off); off = al(off + (size_t)FACTOR * E * sizeof(float));
  float* dcol = (float*)(ws + off); off = al(off + (size_t)FACTOR * N * sizeof(float));
  float* femb = (float*)(ws + off); off = al(off + (size_t)N * EMBED * sizeof(float));
  (void)ws_size; (void)n_in; (void)out_size;

  const int B = 256;
  auto cdiv = [](long long a, long long b) { return (int)((a + b - 1) / b); };

  k_fill<<<cdiv((long long)FACTOR * E, B), B, 0, stream>>>(fv, 1.0f, (long long)FACTOR * E);

  for (int it = 0; it < 2; ++it) {
    k_softmax<<<cdiv(E, B), B, 0, stream>>>(fv, sc, E);

    k_fill<<<cdiv((long long)FACTOR * N, B), B, 0, stream>>>(dcol, 0.0f, (long long)FACTOR * N);
    dim3 gb(cdiv(E, B), FACTOR);
    k_colsum<<<gb, B, 0, stream>>>(head, sc, dcol, E, N);
    k_invsqrt<<<cdiv((long long)FACTOR * N, B), B, 0, stream>>>(dcol, (long long)FACTOR * N);

    k_fill<<<cdiv((long long)N * EMBED, B), B, 0, stream>>>(femb, 0.0f, (long long)N * EMBED);
    k_scatter_msg<<<cdiv(E, DT), 256, 0, stream>>>(ue, ie, n_user, head, tail, E, N, sc, dcol, femb);
    k_scale<<<cdiv((long long)N * EMBED, B), B, 0, stream>>>(femb, dcol, N);

    if (it == 0)  // last iteration's fv update does not affect the output
      k_update_fv<<<cdiv(E, 16), 32, 0, stream>>>(ue, ie, n_user, head, tail, E, N, femb, fv);
  }

  k_out<<<cdiv((long long)N * EMBED, B), B, 0, stream>>>(ue, ie, n_user, femb, (float*)d_out, N);
}
